// MultiHeadAttention_6158983102770
// MI455X (gfx1250) — compile-verified, matrix operand order checked by reading, not run on hardware
//
#include <hip/hip_runtime.h>
#include <hip/hip_bf16.h>

// ---- types ----
typedef __attribute__((ext_vector_type(16))) __bf16 v16bf;
typedef __attribute__((ext_vector_type(8)))  __bf16 v8bf;
typedef __attribute__((ext_vector_type(4)))  __bf16 v4bf;
typedef __attribute__((ext_vector_type(8)))  float  v8f;
typedef unsigned int u32x4 __attribute__((ext_vector_type(4)));
typedef int          i32x4 __attribute__((ext_vector_type(4)));
typedef int          i32x8 __attribute__((ext_vector_type(8)));

#define WMMA_BF16(A_, B_, C_) \
  __builtin_amdgcn_wmma_f32_16x16x32_bf16(false, (A_), false, (B_), (short)0, (C_), false, false)

#if __has_builtin(__builtin_amdgcn_tensor_load_to_lds) && \
    __has_builtin(__builtin_amdgcn_s_wait_tensorcnt)
#define HAVE_TDM 1
#else
#define HAVE_TDM 0
#endif

// Problem constants (reference: B=2, T=2048, C=1024, H=16, D=64)
#define NB 2
#define NT 2048
#define NC 1024
#define NH 16
#define ND 64
#define NM (NB * NT)   // 4096 rows

// ---------------------------------------------------------------------------
// Fragment loaders. CDNA5 16-bit A-matrix 16x32 layout (CDNA5 ISA):
//   lane L: m = L&15, half = L>>4
//   elements 0..7  -> K = half*8 + j        (16 contiguous bytes)
//   elements 8..15 -> K = 16 + half*8 + j   (16 contiguous bytes, +32B away)
// Works for both global and LDS pointers (LDS -> ds_load_b128 pairs).
// ---------------------------------------------------------------------------
__device__ __forceinline__ v16bf load_frag16x32(const __bf16* base, int ld) {
  const int lane = threadIdx.x & 31;
  const int m    = lane & 15;
  const int hf   = lane >> 4;
  const __bf16* p = base + (size_t)m * (size_t)ld + hf * 8;
  v8bf lo = *(const v8bf*)(p);
  v8bf hi = *(const v8bf*)(p + 16);
  v16bf r;
#pragma unroll
  for (int j = 0; j < 8; ++j) { r[j] = lo[j]; r[j + 8] = hi[j]; }
  return r;
}

// Same but only K=0..15 valid (K=16..31 zeroed) — used for the 16-deep P*V step.
__device__ __forceinline__ v16bf load_frag16x32_lo(const __bf16* __restrict__ base, int ld) {
  const int lane = threadIdx.x & 31;
  const int m    = lane & 15;
  const int hf   = lane >> 4;
  v8bf lo = *(const v8bf*)(base + (size_t)m * (size_t)ld + hf * 8);
  v16bf r;
#pragma unroll
  for (int j = 0; j < 8; ++j) { r[j] = lo[j]; r[j + 8] = (__bf16)0.0f; }
  return r;
}

// ---------------------------------------------------------------------------
// Tensor Data Mover: issue a 2D tile load (tile_d0=32 bf16 per row, `rows`
// rows, row stride `strideElems`) from global into LDS at byte addr lds_addr.
// D# packing per CDNA5 ISA:
//   g0: count=1 | lds_addr | global_addr[56:0] | type=2
//   g1: data_size=2B, tensor_dim0/1 == tile_dim0/1 (tile fully in bounds),
//       tensor_dim0_stride = strideElems (data_size units)
// ---------------------------------------------------------------------------
#if HAVE_TDM
__device__ __forceinline__ unsigned lds_off(const void* p) {
  return (unsigned)(size_t)p;   // LDS aperture: low 32 bits are the LDS byte address
}

__device__ __forceinline__ void tdm_load_2d(unsigned lds_addr, const __bf16* gptr,
                                            int rows, long strideElems) {
  unsigned long long ga = (unsigned long long)(size_t)gptr;
  const unsigned td0 = 32u;                  // tile/tensor dim0 (elements per row)
  const unsigned td1 = (unsigned)rows;       // tile/tensor dim1 (rows)
  const unsigned long long s0 = (unsigned long long)strideElems;

  u32x4 g0;
  g0[0] = 1u;                                               // count=1, flags=0
  g0[1] = lds_addr;                                         // lds_addr
  g0[2] = (unsigned)ga;                                     // global_addr[31:0]
  g0[3] = (unsigned)((ga >> 32) & 0x1FFFFFFull) | (2u << 30); // addr[56:32] | type=2

  i32x8 g1;
  g1[0] = (int)(1u << 16);                                  // data_size code 1 = 2 bytes
  g1[1] = (int)((td0 & 0xFFFFu) << 16);                     // tensor_dim0 lo16
  g1[2] = (int)(((td0 >> 16) & 0xFFFFu) | ((td1 & 0xFFFFu) << 16)); // dim0 hi | dim1 lo
  g1[3] = (int)(((td1 >> 16) & 0xFFFFu) | ((td0 & 0xFFFFu) << 16)); // dim1 hi | tile_dim0
  g1[4] = (int)(td1 & 0xFFFFu);                             // tile_dim1 (tile_dim2=0)
  g1[5] = (int)(unsigned)(s0 & 0xFFFFFFFFull);              // dim0_stride lo32
  g1[6] = (int)(unsigned)((s0 >> 32) & 0xFFFFull);          // dim0_stride hi16 (dim1_stride=0)
  g1[7] = 0;

  i32x4 z4 = {0, 0, 0, 0};
#if defined(__clang_major__) && (__clang_major__ >= 23)
  i32x8 z8 = {0, 0, 0, 0, 0, 0, 0, 0};
  __builtin_amdgcn_tensor_load_to_lds(g0, g1, z4, z4, z8, 0);
#else
  __builtin_amdgcn_tensor_load_to_lds(g0, g1, z4, z4, 0);
#endif
}
#endif // HAVE_TDM

// ---------------------------------------------------------------------------
// Shared GEMM machinery: block = 4 waves (2x2), block tile 64x128,
// wave tile 32x64 (2x4 WMMA accumulators), K staged in 32-deep LDS chunks.
// ---------------------------------------------------------------------------
__device__ __forceinline__ void gemm_chunk(const __bf16* xs, const __bf16* ws,
                                           int wm, int wn, v8f acc[2][4]) {
  v16bf a0 = load_frag16x32(xs + (wm * 32) * 32, 32);
  v16bf a1 = load_frag16x32(xs + (wm * 32 + 16) * 32, 32);
  v16bf bfr[4];
#pragma unroll
  for (int j = 0; j < 4; ++j)
    bfr[j] = load_frag16x32(ws + (wn * 64 + j * 16) * 32, 32);
#pragma unroll
  for (int j = 0; j < 4; ++j) {
    acc[0][j] = WMMA_BF16(a0, bfr[j], acc[0][j]);
    acc[1][j] = WMMA_BF16(a1, bfr[j], acc[1][j]);
  }
}

// Ablk: 64 rows of A for this block; Bblk: 128 rows of B^T. Both ld = NC.
// xs/ws: double-buffered LDS (xs: 2*64*32, ws: 2*128*32 bf16).
__device__ __forceinline__ void gemm_mainloop(const __bf16* __restrict__ Ablk,
                                              const __bf16* __restrict__ Bblk,
                                              __bf16* xs, __bf16* ws,
                                              int wm, int wn, v8f acc[2][4]) {
  const int NCH = NC / 32;
#if HAVE_TDM
  const bool issuer = (threadIdx.x >> 5) == 0;   // wave 0 drives the TDM
  if (issuer) {
    tdm_load_2d(lds_off(xs), Ablk, 64, NC);
    tdm_load_2d(lds_off(ws), Bblk, 128, NC);
  }
  for (int kc = 0; kc < NCH; ++kc) {
    const int cur = kc & 1;
    if (kc + 1 < NCH) {
      if (issuer) {
        tdm_load_2d(lds_off(xs + (cur ^ 1) * 64 * 32), Ablk + (kc + 1) * 32, 64, NC);
        tdm_load_2d(lds_off(ws + (cur ^ 1) * 128 * 32), Bblk + (kc + 1) * 32, 128, NC);
        __builtin_amdgcn_s_wait_tensorcnt(2);   // current buffer's pair complete
      }
    } else if (issuer) {
      __builtin_amdgcn_s_wait_tensorcnt(0);
    }
    __syncthreads();
    gemm_chunk(xs + cur * 64 * 32, ws + cur * 128 * 32, wm, wn, acc);
    __syncthreads();                            // safe to overwrite this buffer
  }
#else
  // Cooperative staging fallback (single-buffered)
  for (int kc = 0; kc < NCH; ++kc) {
    const int k0 = kc * 32;
    __syncthreads();
    for (int i = threadIdx.x; i < 64 * 4; i += 128) {
      int row = i >> 2, seg = i & 3;
      *(v8bf*)(xs + row * 32 + seg * 8) = *(const v8bf*)(Ablk + (size_t)row * NC + k0 + seg * 8);
    }
    for (int i = threadIdx.x; i < 128 * 4; i += 128) {
      int row = i >> 2, seg = i & 3;
      *(v8bf*)(ws + row * 32 + seg * 8) = *(const v8bf*)(Bblk + (size_t)row * NC + k0 + seg * 8);
    }
    __syncthreads();
    gemm_chunk(xs, ws, wm, wn, acc);
  }
#endif
}

// ---------------------------------------------------------------------------
// Prep kernels
// ---------------------------------------------------------------------------
__global__ void k_f32_to_bf16(const float* __restrict__ in, __bf16* __restrict__ out, int n) {
  int i = (blockIdx.x * blockDim.x + threadIdx.x) * 4;
  if (i + 3 < n) {
    float4 v = *(const float4*)(in + i);
    v4bf o;
    o[0] = (__bf16)v.x; o[1] = (__bf16)v.y; o[2] = (__bf16)v.z; o[3] = (__bf16)v.w;
    *(v4bf*)(out + i) = o;
  }
}

// W[K,N] f32 -> Wt[N,K] bf16 (compile-time N => divisions become shifts)
template <int N>
__global__ void k_transpose_bf16(const float* __restrict__ W, __bf16* __restrict__ Wt, int K) {
  int idx = blockIdx.x * blockDim.x + threadIdx.x;
  if (idx < K * N) {
    int k = idx / N;
    int n = idx - k * N;
    Wt[(size_t)n * K + k] = (__bf16)W[idx];
  }
}

// ---------------------------------------------------------------------------
// QKV GEMM: qkv = x(4096x1024) * Wqkv(1024x3072) + b, scattered to
// q[B,H,T,D], k[B,H,T,D], vT[B,H,D,T] in bf16.
// ---------------------------------------------------------------------------
__global__ void k_gemm_qkv(const __bf16* __restrict__ X, const __bf16* __restrict__ Wt,
                           const float* __restrict__ bias,
                           __bf16* __restrict__ qb, __bf16* __restrict__ kb,
                           __bf16* __restrict__ vtb) {
  __shared__ __bf16 Xs[2][64 * 32];
  __shared__ __bf16 Ws[2][128 * 32];

  const int lane = threadIdx.x & 31;
  const int wave = threadIdx.x >> 5;
  const int wm = wave & 1, wn = wave >> 1;
  const int bm = blockIdx.x * 64 + wm * 32;
  const int bn = blockIdx.y * 128 + wn * 64;

  v8f acc[2][4];
#pragma unroll
  for (int i = 0; i < 2; ++i)
#pragma unroll
    for (int j = 0; j < 4; ++j)
#pragma unroll
      for (int r = 0; r < 8; ++r) acc[i][j][r] = 0.0f;

  gemm_mainloop(X + (size_t)(blockIdx.x * 64) * NC,
                Wt + (size_t)(blockIdx.y * 128) * NC,
                &Xs[0][0], &Ws[0][0], wm, wn, acc);

  const int hf = lane >> 4, nl = lane & 15;
#pragma unroll
  for (int i = 0; i < 2; ++i) {
#pragma unroll
    for (int j = 0; j < 4; ++j) {
#pragma unroll
      for (int r = 0; r < 8; ++r) {
        int m   = bm + i * 16 + r + hf * 8;
        int col = bn + j * 16 + nl;
        float v = acc[i][j][r] + bias[col];
        int b = m >> 11, t = m & (NT - 1);
        int sec = col >> 10, c = col & (NC - 1);
        int h = c >> 6, d = c & (ND - 1);
        __bf16 bv = (__bf16)v;
        size_t bh = (size_t)(b * NH + h);
        if (sec == 0)      qb[(bh * NT + t) * ND + d]  = bv;
        else if (sec == 1) kb[(bh * NT + t) * ND + d]  = bv;
        else               vtb[(bh * ND + d) * NT + t] = bv;
      }
    }
  }
}

// ---------------------------------------------------------------------------
// Flash attention. One wave per (b, h, 16-query tile). Computes S^T = K*Q^T so
// each lane holds a fixed query with 8 key values in registers — exactly the
// A-fragment layout needed for P in O = P*V (V pre-transposed to [D,T]).
// ---------------------------------------------------------------------------
__global__ void k_attn(const __bf16* __restrict__ q, const __bf16* __restrict__ k,
                       const __bf16* __restrict__ vt, __bf16* __restrict__ att) {
  const int lane = threadIdx.x & 31;
  const int wave = threadIdx.x >> 5;
  const int h = blockIdx.y, b = blockIdx.z;
  const int qbase = (blockIdx.x * 4 + wave) * 16;
  const size_t bh = (size_t)(b * NH + h);

  const __bf16* qh = q  + bh * NT * ND;
  const __bf16* kh = k  + bh * NT * ND;
  const __bf16* vh = vt + bh * ND * NT;

  const int nl = lane & 15, hf = lane >> 4;

  v16bf qf0 = load_frag16x32(qh + (size_t)qbase * ND,      ND);
  v16bf qf1 = load_frag16x32(qh + (size_t)qbase * ND + 32, ND);

  v8f o[4];
#pragma unroll
  for (int n = 0; n < 4; ++n)
#pragma unroll
    for (int r = 0; r < 8; ++r) o[n][r] = 0.0f;

  float m_run = -1e30f, l_run = 0.0f;
  const int qi = qbase + nl;

  for (int kb0 = 0; kb0 <= qbase; kb0 += 16) {
    v16bf kf0 = load_frag16x32(kh + (size_t)kb0 * ND,      ND);
    v16bf kf1 = load_frag16x32(kh + (size_t)kb0 * ND + 32, ND);
    v8f z;
#pragma unroll
    for (int r = 0; r < 8; ++r) z[r] = 0.0f;
    v8f st = WMMA_BF16(kf1, qf1, WMMA_BF16(kf0, qf0, z));
    // st[r]: S^T[key = kb0 + r + hf*8][query = qbase + nl]

    float s[8];
    float mx = -1e30f;
#pragma unroll
    for (int r = 0; r < 8; ++r) {
      int key  = kb0 + r + hf * 8;
      float sv = st[r] * 0.125f;              // 1/sqrt(64)
      if (key > qi) sv = -1e30f;              // causal mask
      s[r] = sv;
      mx = fmaxf(mx, sv);
    }
    mx = fmaxf(mx, __shfl_xor(mx, 16));       // combine both key halves of this query
    float m_new = fmaxf(m_run, mx);
    float alpha = __expf(m_run - m_new);

    v16bf pf;
    float psum = 0.0f;
#pragma unroll
    for (int r = 0; r < 8; ++r) {
      float p = __expf(s[r] - m_new);
      psum += p;
      pf[r] = (__bf16)p;
      pf[r + 8] = (__bf16)0.0f;               // K=16..31 padding
    }
    l_run = l_run * alpha + psum;
    m_run = m_new;

    float ar[8];
#pragma unroll
    for (int r = 0; r < 8; ++r)
      ar[r] = __shfl(alpha, r + hf * 8);      // alpha for output row m = r + hf*8

#pragma unroll
    for (int nt = 0; nt < 4; ++nt) {
      v16bf vf = load_frag16x32_lo(vh + (size_t)(nt * 16) * NT + kb0, NT);
      v8f c = o[nt];
#pragma unroll
      for (int r = 0; r < 8; ++r) c[r] *= ar[r];
      o[nt] = WMMA_BF16(pf, vf, c);
    }
  }

  float l_tot = l_run + __shfl_xor(l_run, 16);
  float lr[8];
#pragma unroll
  for (int r = 0; r < 8; ++r)
    lr[r] = 1.0f / __shfl(l_tot, r + hf * 8);

  __bf16* outp = att + ((size_t)b * NT + qbase) * NC + h * ND;
#pragma unroll
  for (int nt = 0; nt < 4; ++nt)
#pragma unroll
    for (int r = 0; r < 8; ++r)
      outp[(size_t)(r + hf * 8) * NC + nt * 16 + nl] = (__bf16)(o[nt][r] * lr[r]);
}

// ---------------------------------------------------------------------------
// Output projection: out = att(4096x1024) * Wproj(1024x1024) + b  (f32 out)
// ---------------------------------------------------------------------------
__global__ void k_gemm_proj(const __bf16* __restrict__ A, const __bf16* __restrict__ Wt,
                            const float* __restrict__ bias, float* __restrict__ out) {
  __shared__ __bf16 Xs[2][64 * 32];
  __shared__ __bf16 Ws[2][128 * 32];

  const int lane = threadIdx.x & 31;
  const int wave = threadIdx.x >> 5;
  const int wm = wave & 1, wn = wave >> 1;
  const int bm = blockIdx.x * 64 + wm * 32;
  const int bn = blockIdx.y * 128 + wn * 64;

  v8f acc[2][4];
#pragma unroll
  for (int i = 0; i < 2; ++i)
#pragma unroll
    for (int j = 0; j < 4; ++j)
#pragma unroll
      for (int r = 0; r < 8; ++r) acc[i][j][r] = 0.0f;

  gemm_mainloop(A + (size_t)(blockIdx.x * 64) * NC,
                Wt + (size_t)(blockIdx.y * 128) * NC,
                &Xs[0][0], &Ws[0][0], wm, wn, acc);

  const int hf = lane >> 4, nl = lane & 15;
#pragma unroll
  for (int i = 0; i < 2; ++i)
#pragma unroll
    for (int j = 0; j < 4; ++j)
#pragma unroll
      for (int r = 0; r < 8; ++r) {
        int m   = bm + i * 16 + r + hf * 8;
        int col = bn + j * 16 + nl;
        out[(size_t)m * NC + col] = acc[i][j][r] + bias[col];
      }
}

// ---------------------------------------------------------------------------
// Host launcher
// ---------------------------------------------------------------------------
extern "C" void kernel_launch(void* const* d_in, const int* in_sizes, int n_in,
                              void* d_out, int out_size, void* d_ws, size_t ws_size,
                              hipStream_t stream) {
  const float* x     = (const float*)d_in[0];   // [B,T,C]
  const float* Wqkv  = (const float*)d_in[1];   // [C,3C]
  const float* bqkv  = (const float*)d_in[2];   // [3C]
  const float* Wproj = (const float*)d_in[3];   // [C,C]
  const float* bproj = (const float*)d_in[4];   // [C]
  float* out = (float*)d_out;                   // [B,T,C] f32

  // Workspace layout (bf16), ~48 MiB total
  __bf16* xb     = (__bf16*)d_ws;                  // 4096*1024
  __bf16* wqkvt  = xb     + (size_t)NM * NC;       // 3072*1024
  __bf16* wprojt = wqkvt  + (size_t)3 * NC * NC;   // 1024*1024
  __bf16* qb     = wprojt + (size_t)NC * NC;       // B*H*T*D
  __bf16* kbuf   = qb     + (size_t)NM * NC;
  __bf16* vtb    = kbuf   + (size_t)NM * NC;
  __bf16* attb   = vtb    + (size_t)NM * NC;

  // 1) Convert x to bf16
  k_f32_to_bf16<<<(NM * NC) / (256 * 4), 256, 0, stream>>>(x, xb, NM * NC);
  // 2) Transpose+convert weights to B^T bf16
  k_transpose_bf16<3 * NC><<<(3 * NC * NC + 255) / 256, 256, 0, stream>>>(Wqkv, wqkvt, NC);
  k_transpose_bf16<NC><<<(NC * NC + 255) / 256, 256, 0, stream>>>(Wproj, wprojt, NC);
  // 3) QKV projection (TDM -> LDS -> WMMA) -> q, k, v^T
  k_gemm_qkv<<<dim3(NM / 64, (3 * NC) / 128), 128, 0, stream>>>(xb, wqkvt, bqkv, qb, kbuf, vtb);
  // 4) Causal flash attention (WMMA)
  k_attn<<<dim3(NT / 64, NH, NB), 128, 0, stream>>>(qb, kbuf, vtb, attb);
  // 5) Output projection (TDM -> LDS -> WMMA) -> f32 out
  k_gemm_proj<<<dim3(NM / 64, NC / 128), 128, 0, stream>>>(attb, wprojt, bproj, out);
}
